// MambaRegressor_34797825032736
// MI455X (gfx1250) — matrix operand order repaired and checked by reading, not compiled, not run on hardware
//
#include <hip/hip_runtime.h>
#include <hip/hip_bf16.h>

// ---------------- types ----------------
typedef __attribute__((ext_vector_type(16))) _Float16 v16h;
typedef __attribute__((ext_vector_type(8)))  _Float16 v8h;
typedef __attribute__((ext_vector_type(8)))  float    v8f;

// ---------------- model dims ----------------
#define B_      128
#define K_      128
#define DIN     256
#define PROJ    64
#define DMODEL  128
#define NLAYER  4
#define PLEN    8
#define NPATCH  32
#define DINNER  256
#define DSTATE  16
#define DTRANK  8
#define DCONV   4
#define BH      (B_*PROJ)          // 8192
#define NTOK    (BH*NPATCH)        // 262144 tokens

// ======================================================================
// Generic WMMA GEMM:  C[M,N] = A[M,K] @ Bw[N,K]^T  (+bias[N]) (+resid[M,ldc])
// A,Bw are f16 row-major.  One wave computes a 16 x (16*NBT) tile.
// Uses v_wmma_f32_16x16x32_f16 (wave32, f32 accumulate).
// ======================================================================
template<int NBT, typename TC>
__global__ __launch_bounds__(256) void gemm_wmma(
    const _Float16* __restrict__ A, const _Float16* __restrict__ Bw,
    TC* __restrict__ C, const float* __restrict__ bias,
    const float* __restrict__ resid,
    int M, int N, int K, int ldc, int doClamp, float clo, float chi)
{
    const int lane = threadIdx.x & 31;
    const int wave = threadIdx.x >> 5;
    const int gw   = blockIdx.x * 8 + wave;
    const int ntb  = N / (16 * NBT);
    const int total = (M >> 4) * ntb;
    if (gw >= total) return;
    const int m0 = (gw / ntb) << 4;
    const int n0 = (gw % ntb) * (16 * NBT);
    const int r   = lane & 15;     // row/col index within 16
    const int hi  = lane >> 4;     // half-wave select
    const int col = lane & 15;     // D-matrix column (N)
    const int rb  = m0 + hi * 8;   // D-matrix row base (M)

    v8f acc[NBT];
    #pragma unroll
    for (int t = 0; t < NBT; ++t) {
        #pragma unroll
        for (int v = 0; v < 8; ++v) {
            float c0 = 0.f;
            if (bias)  c0 += bias[n0 + t * 16 + col];
            if (resid) c0 += resid[(size_t)(rb + v) * ldc + n0 + t * 16 + col];
            acc[t][v] = c0;
        }
    }

    // A fragment: lane<16 -> M=m0+r, K k0+0..15 contiguous
    //             lane>=16 -> M=m0+r, K k0+16..31
    const _Float16* ap = A + (size_t)(m0 + r) * K + hi * 16;
    for (int k0 = 0; k0 < K; k0 += 32) {
        __builtin_prefetch(ap + k0 + 32, 0, 1);
        v8h alo = *(const v8h*)(ap + k0);
        v8h ahi = *(const v8h*)(ap + k0 + 8);
        v16h a = __builtin_shufflevector(alo, ahi,
                    0,1,2,3,4,5,6,7,8,9,10,11,12,13,14,15);
        #pragma unroll
        for (int t = 0; t < NBT; ++t) {
            // B fragment: lane<16 -> N=n0+t*16+r, K k0+0..15 contiguous
            //             lane>=16 -> same N, K k0+16..31
            const _Float16* bp = Bw + (size_t)(n0 + t * 16 + r) * K + k0 + hi * 16;
            v16h b = *(const v16h*)bp;
            acc[t] = __builtin_amdgcn_wmma_f32_16x16x32_f16(
                        false, a, false, b, (short)0, acc[t], false, false);
        }
    }

    #pragma unroll
    for (int t = 0; t < NBT; ++t) {
        #pragma unroll
        for (int v = 0; v < 8; ++v) {
            float val = acc[t][v];
            if (doClamp) val = fminf(fmaxf(val, clo), chi);
            C[(size_t)(rb + v) * ldc + n0 + t * 16 + col] = (TC)val;
        }
    }
}

// ---------------- helpers ----------------
__device__ __forceinline__ float block_reduce_sum(float v, float* red)
{
    int tid = threadIdx.x;
    red[tid] = v; __syncthreads();
    for (int s = 128; s > 0; s >>= 1) {
        if (tid < s) red[tid] += red[tid + s];
        __syncthreads();
    }
    float r = red[0]; __syncthreads();
    return r;
}

// f32 -> f16 copy
__global__ __launch_bounds__(256) void cvt_f16_kernel(
    const float* __restrict__ in, _Float16* __restrict__ out, size_t n)
{
    size_t i = (size_t)blockIdx.x * 256 + threadIdx.x;
    if (i < n) out[i] = (_Float16)in[i];
}

// pad xproj_w (NLAYER,40,256) -> f16 (NLAYER,48,256), zero rows 40..47
__global__ __launch_bounds__(256) void xproj_pad_kernel(
    const float* __restrict__ w, _Float16* __restrict__ o)
{
    int i = blockIdx.x * 256 + threadIdx.x;
    if (i >= NLAYER * 48 * 256) return;
    int layer = i / (48 * 256);
    int rem   = i % (48 * 256);
    int row = rem / 256, colc = rem % 256;
    float v = (row < 40) ? w[layer * 40 * 256 + row * 256 + colc] : 0.f;
    o[i] = (_Float16)v;
}

__global__ __launch_bounds__(256) void zero_kernel(float* p, int n)
{
    int i = blockIdx.x * 256 + threadIdx.x;
    if (i < n) p[i] = 0.f;
}

// sum |x| over n -> atomicAdd into out[0]
__global__ __launch_bounds__(256) void abssum_kernel(
    const float* __restrict__ x, float* __restrict__ out, size_t n)
{
    __shared__ float red[256];
    float s = 0.f;
    for (size_t i = (size_t)blockIdx.x * 256 + threadIdx.x; i < n;
         i += (size_t)gridDim.x * 256)
        s += fabsf(x[i]);
    float r = block_reduce_sum(s, red);
    if (threadIdx.x == 0) atomicAdd(out, r);
}

// RevIN stats + replication-pad patching + patch embedding.
// One block per (b,c). xp layout (B,K,PROJ).
__global__ __launch_bounds__(256) void patch_embed_kernel(
    const float* __restrict__ xp, const float* __restrict__ sumabs,
    const float* __restrict__ emb_w, const float* __restrict__ emb_b,
    float* __restrict__ h, float* __restrict__ msd)
{
    int bc = blockIdx.x;                 // b*64 + c
    int b = bc >> 6, c = bc & 63;
    int tid = threadIdx.x;
    __shared__ float xn[K_];
    __shared__ float red[256];
    __shared__ float se[DMODEL * PLEN];  // emb_w
    __shared__ float eb[DMODEL];

    float scale = 1.f / (sumabs[0] * (1.f / (float)(B_ * K_ * PROJ)) + 1e-6f);
    float v = 0.f;
    if (tid < K_) v = xp[((size_t)b * K_ + tid) * PROJ + c] * scale;

    float mu = block_reduce_sum((tid < K_) ? v : 0.f, red) * (1.f / (float)K_);
    float dv = (tid < K_) ? (v - mu) : 0.f;
    float var = block_reduce_sum(dv * dv, red) * (1.f / (float)K_);
    float sd = sqrtf(var + 1e-5f);
    if (tid == 0) { msd[bc * 2 + 0] = mu; msd[bc * 2 + 1] = sd; }
    if (tid < K_) xn[tid] = (v - mu) / sd;

    // stage embedding weights
    for (int i = tid; i < DMODEL * PLEN; i += 256) se[i] = emb_w[i];
    if (tid < DMODEL) eb[tid] = emb_b[tid];
    __syncthreads();

    // h[bc, n, dm] = sum_p patch[n,p] * emb_w[dm,p] + emb_b[dm]
    for (int k = 0; k < (NPATCH * DMODEL) / 256; ++k) {
        int o = k * 256 + tid;           // n*128 + dm
        int n = o >> 7, dm = o & 127;
        float acc = eb[dm];
        #pragma unroll
        for (int p = 0; p < PLEN; ++p) {
            int kk = n * 4 + p;          // replication pad == clamp to last
            if (kk > K_ - 1) kk = K_ - 1;
            acc += xn[kk] * se[dm * PLEN + p];
        }
        h[(size_t)bc * (NPATCH * DMODEL) + o] = acc;
    }
}

// rmsnorm over d=128 per token; one wave per token, 8 waves/block; f16 out
__global__ __launch_bounds__(256) void rmsnorm_kernel(
    const float* __restrict__ h, const float* __restrict__ w,
    _Float16* __restrict__ out, int ntok)
{
    int tok = blockIdx.x * 8 + (threadIdx.x >> 5);
    if (tok >= ntok) return;
    int lane = threadIdx.x & 31;
    const float* row = h + (size_t)tok * DMODEL;
    float v[4], s = 0.f;
    #pragma unroll
    for (int i = 0; i < 4; ++i) { v[i] = row[lane + i * 32]; s += v[i] * v[i]; }
    #pragma unroll
    for (int off = 16; off > 0; off >>= 1) s += __shfl_xor(s, off, 32);
    float r = rsqrtf(s * (1.f / (float)DMODEL) + 1e-5f);
    #pragma unroll
    for (int i = 0; i < 4; ++i)
        out[(size_t)tok * DMODEL + lane + i * 32] =
            (_Float16)(v[i] * r * w[lane + i * 32]);
}

// depthwise causal conv (k=4, left pad 3) + bias + silu
// xz: (BH, L, 512) f16 (xc = cols 0..255). xc_out: (BH, L, 256) f16
__global__ __launch_bounds__(256) void conv_silu_kernel(
    const _Float16* __restrict__ xz, _Float16* __restrict__ xc,
    const float* __restrict__ cw, const float* __restrict__ cb)
{
    size_t idx = (size_t)blockIdx.x * 256 + threadIdx.x;  // BH*L*256
    int d = (int)(idx & 255);
    size_t t = idx >> 8;                 // bh*32 + l
    int l = (int)(t & 31);
    size_t base = (t - (size_t)l) * 512; // bh*L*512
    float acc = cb[d];
    #pragma unroll
    for (int j = 0; j < DCONV; ++j) {
        int ls = l - (DCONV - 1) + j;
        if (ls >= 0)
            acc += cw[d * DCONV + j] * (float)xz[base + (size_t)ls * 512 + d];
    }
    float s = acc / (1.f + __expf(-acc));
    xc[idx] = (_Float16)s;
}

// Fused dtproj + softplus + selective scan + D skip + silu(res) gate.
// One block per bh (256 threads = one d_inner channel each). y written
// in place over xc (u buffer) -> becomes A operand of outproj GEMM.
__global__ __launch_bounds__(256) void scan_kernel(
    const _Float16* __restrict__ xz,   // (BH,L,512), res at +256
    _Float16* __restrict__ xc,         // (BH,L,256) in: u, out: y*silu(res)
    const float* __restrict__ xpo,     // (BH,L,48): dt[0:8] B[8:24] C[24:40]
    const float* __restrict__ dtw,     // (256,8)
    const float* __restrict__ dtb,     // (256)
    const float* __restrict__ A_log,   // (256,16)
    const float* __restrict__ Dp)      // (256)
{
    int bh = blockIdx.x;
    int d  = threadIdx.x;
    __shared__ float sb[48];

    float w[DTRANK];
    #pragma unroll
    for (int j = 0; j < DTRANK; ++j) w[j] = dtw[d * DTRANK + j];
    float bias = dtb[d];
    float Ar[DSTATE];
    #pragma unroll
    for (int n = 0; n < DSTATE; ++n) Ar[n] = -__expf(A_log[d * DSTATE + n]);
    float Dd = Dp[d];
    float hst[DSTATE];
    #pragma unroll
    for (int n = 0; n < DSTATE; ++n) hst[n] = 0.f;

    const size_t ub = (size_t)bh * NPATCH * DINNER;
    const size_t zb = (size_t)bh * NPATCH * 512;
    const size_t pb = (size_t)bh * NPATCH * 48;

    for (int l = 0; l < NPATCH; ++l) {
        if (d < 48) sb[d] = xpo[pb + (size_t)l * 48 + d];
        __syncthreads();
        float dt = bias;
        #pragma unroll
        for (int j = 0; j < DTRANK; ++j) dt += sb[j] * w[j];
        float delta = (dt > 20.f) ? dt : log1pf(__expf(dt));  // softplus
        float u = (float)xc[ub + (size_t)l * DINNER + d];
        float y = 0.f;
        #pragma unroll
        for (int n = 0; n < DSTATE; ++n) {
            float dA = __expf(delta * Ar[n]);
            hst[n] = hst[n] * dA + delta * sb[8 + n] * u;
            y += hst[n] * sb[24 + n];
        }
        y += u * Dd;
        float res = (float)xz[zb + (size_t)l * 512 + DINNER + d];
        float g = res / (1.f + __expf(-res));                 // silu
        __syncthreads();
        xc[ub + (size_t)l * DINNER + d] = (_Float16)(y * g);
    }
}

// head GEMV per bh row (4096-wide) + RevIN denorm
__global__ __launch_bounds__(256) void head_gemv_kernel(
    const _Float16* __restrict__ hf, const float* __restrict__ hw,
    const float* __restrict__ hb, const float* __restrict__ msd,
    float* __restrict__ ybc)
{
    __shared__ float red[256];
    int bh = blockIdx.x;
    float s = 0.f;
    for (int i = threadIdx.x; i < NPATCH * DMODEL; i += 256)
        s += (float)hf[(size_t)bh * (NPATCH * DMODEL) + i] * hw[i];
    float r = block_reduce_sum(s, red);
    if (threadIdx.x == 0) {
        float y = r + hb[0];
        ybc[bh] = y * msd[bh * 2 + 1] + msd[bh * 2 + 0];
    }
}

// final head: out[b,j] = sum_c ybc[b*64+c]*head_w[j,c] + head_b[j]
__global__ __launch_bounds__(256) void final_head_kernel(
    const float* __restrict__ ybc, const float* __restrict__ hw,
    const float* __restrict__ hb, float* __restrict__ out)
{
    int t = threadIdx.x;               // 256 = 128*2
    int b = t >> 1, j = t & 1;
    float acc = hb[j];
    #pragma unroll 4
    for (int c = 0; c < PROJ; ++c) acc += ybc[b * PROJ + c] * hw[j * PROJ + c];
    out[b * 2 + j] = acc;
}

// ======================================================================
extern "C" void kernel_launch(void* const* d_in, const int* in_sizes, int n_in,
                              void* d_out, int out_size, void* d_ws, size_t ws_size,
                              hipStream_t stream)
{
    (void)in_sizes; (void)n_in; (void)out_size; (void)ws_size;
    const float* x        = (const float*)d_in[0];
    const float* proj_w   = (const float*)d_in[1];
    const float* proj_b   = (const float*)d_in[2];
    const float* emb_w    = (const float*)d_in[3];
    const float* emb_b    = (const float*)d_in[4];
    const float* norm_w   = (const float*)d_in[5];
    const float* inproj_w = (const float*)d_in[6];
    const float* conv_w   = (const float*)d_in[7];
    const float* conv_b   = (const float*)d_in[8];
    const float* xproj_w  = (const float*)d_in[9];
    const float* dtproj_w = (const float*)d_in[10];
    const float* dtproj_b = (const float*)d_in[11];
    const float* A_log    = (const float*)d_in[12];
    const float* D_p      = (const float*)d_in[13];
    const float* outproj_w= (const float*)d_in[14];
    const float* normf_w  = (const float*)d_in[15];
    const float* headb_w  = (const float*)d_in[16];
    const float* headb_b  = (const float*)d_in[17];
    const float* head_w   = (const float*)d_in[18];
    const float* head_b   = (const float*)d_in[19];
    float* out = (float*)d_out;

    // ---- workspace carve ----
    char* wp = (char*)d_ws;
    auto carve = [&](size_t bytes) -> char* {
        char* p = wp; wp += (bytes + 255) & ~(size_t)255; return p;
    };
    _Float16* x16   = (_Float16*)carve((size_t)B_*K_*DIN * 2);        // 8 MB
    _Float16* pw16  = (_Float16*)carve((size_t)PROJ*DIN * 2);
    float*    xp    = (float*)   carve((size_t)B_*K_*PROJ * 4);       // 4 MB
    float*    sumab = (float*)   carve(256);
    float*    msd   = (float*)   carve((size_t)BH*2 * 4);
    float*    hbuf  = (float*)   carve((size_t)NTOK*DMODEL * 4);      // 128 MB
    _Float16* hn16  = (_Float16*)carve((size_t)NTOK*DMODEL * 2);      // 64 MB
    _Float16* xz16  = (_Float16*)carve((size_t)NTOK*512 * 2);         // 256 MB
    _Float16* xc16  = (_Float16*)carve((size_t)NTOK*DINNER * 2);      // 128 MB
    float*    xpo   = (float*)   carve((size_t)NTOK*48 * 4);          // 48 MB
    _Float16* ipw16 = (_Float16*)carve((size_t)NLAYER*512*DMODEL * 2);
    _Float16* xpw16 = (_Float16*)carve((size_t)NLAYER*48*DINNER * 2);
    _Float16* opw16 = (_Float16*)carve((size_t)NLAYER*DMODEL*DINNER * 2);
    float*    ybc   = (float*)   carve((size_t)BH * 4);

    // ---- weight / input f16 conversion ----
    {
        size_t n = (size_t)B_*K_*DIN;
        cvt_f16_kernel<<<(unsigned)((n + 255) / 256), 256, 0, stream>>>(x, x16, n);
    }
    cvt_f16_kernel<<<(PROJ*DIN + 255) / 256, 256, 0, stream>>>(proj_w, pw16, PROJ*DIN);
    cvt_f16_kernel<<<(NLAYER*512*DMODEL + 255) / 256, 256, 0, stream>>>(
        inproj_w, ipw16, (size_t)NLAYER*512*DMODEL);
    cvt_f16_kernel<<<(NLAYER*DMODEL*DINNER + 255) / 256, 256, 0, stream>>>(
        outproj_w, opw16, (size_t)NLAYER*DMODEL*DINNER);
    xproj_pad_kernel<<<(NLAYER*48*256 + 255) / 256, 256, 0, stream>>>(xproj_w, xpw16);

    // ---- input projection (WMMA) + clamp ----
    {   // M=16384, N=64 (NBT=4 -> 1 n-block), K=256
        int waves = (B_*K_ / 16) * 1;
        gemm_wmma<4, float><<<(waves + 7) / 8, 256, 0, stream>>>(
            x16, pw16, xp, proj_b, nullptr,
            B_*K_, PROJ, DIN, PROJ, 1, -5.f, 5.f);
    }
    // ---- global mean|x| ----
    zero_kernel<<<1, 256, 0, stream>>>(sumab, 1);
    abssum_kernel<<<1024, 256, 0, stream>>>(xp, sumab, (size_t)B_*K_*PROJ);
    // ---- RevIN + patch + embed ----
    patch_embed_kernel<<<BH, 256, 0, stream>>>(xp, sumab, emb_w, emb_b, hbuf, msd);

    // ---- mamba layers ----
    for (int i = 0; i < NLAYER; ++i) {
        rmsnorm_kernel<<<NTOK / 8, 256, 0, stream>>>(
            hbuf, norm_w + i * DMODEL, hn16, NTOK);
        {   // inproj: M=262144, N=512, K=128 -> 8 n-blocks of 64
            int waves = (NTOK / 16) * (512 / 64);
            gemm_wmma<4, _Float16><<<waves / 8, 256, 0, stream>>>(
                hn16, ipw16 + (size_t)i * 512 * DMODEL, xz16,
                nullptr, nullptr, NTOK, 512, DMODEL, 512, 0, 0.f, 0.f);
        }
        conv_silu_kernel<<<(unsigned)((size_t)NTOK * DINNER / 256), 256, 0, stream>>>(
            xz16, xc16, conv_w + i * DINNER * DCONV, conv_b + i * DINNER);
        {   // xproj: M=262144, N=48(padded), K=256 -> NBT=3
            int waves = (NTOK / 16) * 1;
            gemm_wmma<3, float><<<waves / 8, 256, 0, stream>>>(
                xc16, xpw16 + (size_t)i * 48 * DINNER, xpo,
                nullptr, nullptr, NTOK, 48, DINNER, 48, 0, 0.f, 0.f);
        }
        scan_kernel<<<BH, 256, 0, stream>>>(
            xz16, xc16, xpo,
            dtproj_w + i * DINNER * DTRANK, dtproj_b + i * DINNER,
            A_log + i * DINNER * DSTATE, D_p + i * DINNER);
        {   // outproj + residual into hbuf: M=262144, N=128, K=256
            int waves = (NTOK / 16) * (DMODEL / 64);
            gemm_wmma<4, float><<<waves / 8, 256, 0, stream>>>(
                xc16, opw16 + (size_t)i * DMODEL * DINNER, hbuf,
                nullptr, hbuf, NTOK, DMODEL, DINNER, DMODEL, 0, 0.f, 0.f);
        }
    }

    // ---- final norm + heads ----
    rmsnorm_kernel<<<NTOK / 8, 256, 0, stream>>>(hbuf, normf_w, hn16, NTOK);
    head_gemv_kernel<<<BH, 256, 0, stream>>>(hn16, headb_w, headb_b, msd, ybc);
    final_head_kernel<<<1, 256, 0, stream>>>(ybc, head_w, head_b, out);
}
